// TAGConvNet_11940009083384
// MI455X (gfx1250) — matrix operand order repaired and checked by reading, not compiled, not run on hardware
//
#include <hip/hip_runtime.h>
#include <hip/hip_bf16.h>
#include <math.h>

// ---------------- problem constants (match reference) ----------------
static constexpr int N_    = 100000;
static constexpr int E_    = 1600000;
static constexpr int G_    = 64;
static constexpr int KHOP  = 3;
static constexpr int F_IN  = 16;
static constexpr int H_    = 128;
static constexpr int C_    = 40;
static constexpr float EPS = 1e-5f;

typedef __attribute__((ext_vector_type(16))) __bf16 v16bf;
typedef __attribute__((ext_vector_type(8)))  float  v8f;

// ---------------- graph normalization ----------------
__global__ void deg_kernel(const int* __restrict__ col, const float* __restrict__ w,
                           float* __restrict__ deg) {
    int e = blockIdx.x * blockDim.x + threadIdx.x;
    if (e >= E_) return;
    atomicAdd(&deg[col[e]], w[e]);
}

__global__ void norm_kernel(const int* __restrict__ row, const int* __restrict__ col,
                            const float* __restrict__ w, const float* __restrict__ deg,
                            float* __restrict__ norm) {
    int e = blockIdx.x * blockDim.x + threadIdx.x;
    if (e >= E_) return;
    float dr = deg[row[e]], dc = deg[col[e]];
    float ir = dr > 0.f ? rsqrtf(dr) : 0.f;
    float ic = dc > 0.f ? rsqrtf(dc) : 0.f;
    norm[e] = ir * w[e] * ic;
}

__global__ void copy_x_kernel(const float* __restrict__ x, float* __restrict__ xcat1) {
    int idx = blockIdx.x * blockDim.x + threadIdx.x;
    if (idx >= N_ * F_IN) return;
    int n = idx / F_IN, f = idx % F_IN;
    xcat1[(size_t)n * ((KHOP + 1) * F_IN) + f] = x[idx];
}

// out[col] += h[row] * norm  (4 features per thread, float4 load, f32 atomics)
__global__ void propagate_kernel(const float* __restrict__ hin, int ldIn, int offIn,
                                 float* __restrict__ hout, int ldOut, int offOut,
                                 const int* __restrict__ row, const int* __restrict__ col,
                                 const float* __restrict__ norm, int Fq) {
    int idx = blockIdx.x * blockDim.x + threadIdx.x;
    if (idx >= E_ * Fq) return;
    int e = idx / Fq, f4 = idx - e * Fq;
    float w = norm[e];
    const float4 v = *(const float4*)(hin + (size_t)row[e] * ldIn + offIn + f4 * 4);
    float* dst = hout + (size_t)col[e] * ldOut + offOut + f4 * 4;
    atomicAdd(dst + 0, v.x * w);
    atomicAdd(dst + 1, v.y * w);
    atomicAdd(dst + 2, v.z * w);
    atomicAdd(dst + 3, v.w * w);
}

// ---------------- weight pre-pack: f32 [kDim,Nout] -> bf16 WMMA B-fragments ----------------
// out element index = ((colTile*KSTEPS + kstep)*32 + lane)*16 + j
// fragment mapping (16-bit B 32x16): lane -> n = lane&15, h = lane>>4 ; element j<8 -> K = h*8 + j ; j>=8 -> K = 16 + h*8 + (j-8)
__global__ void pack_b_kernel(const float* __restrict__ B, int ldb, int kSteps,
                              __bf16* __restrict__ out, int total) {
    int idx = blockIdx.x * blockDim.x + threadIdx.x;
    if (idx >= total) return;
    int j       = idx & 15;
    int lane    = (idx >> 4) & 31;
    int rest    = idx >> 9;
    int kstep   = rest % kSteps;
    int colTile = rest / kSteps;
    int n = lane & 15, h = lane >> 4;
    int k = kstep * 32 + (j < 8 ? h * 8 + j : 16 + h * 8 + (j - 8));
    out[idx] = (__bf16)B[(size_t)k * ldb + colTile * 16 + n];
}

// ---------------- WMMA GEMM: C[nRows,Nout] = A[nRows,KDIM] @ B ----------------
// block = 256 (8 waves); wave -> one 16x16 C tile. A f32 (cvt on the fly), B pre-packed bf16.
template <int KDIM>
__global__ __launch_bounds__(256)
void gemm_bf16_wmma(const float* __restrict__ A, int lda,
                    const __bf16* __restrict__ Bfrag,
                    float* __restrict__ C, int ldc, int nRows) {
    constexpr int KSTEPS = KDIM / 32;
    const int lane = threadIdx.x & 31;
    const int wave = threadIdx.x >> 5;
    const int rowBase = (blockIdx.x * 8 + wave) * 16;
    if (rowBase >= nRows) return;                 // wave-uniform: EXEC stays all-ones
    const int m = lane & 15;                      // A row within tile / C col
    const int h = lane >> 4;                      // lane half

    const float* arow = A + (size_t)(rowBase + m) * lda + h * 8;
    const v16bf* Bf = (const v16bf*)Bfrag + (size_t)blockIdx.y * KSTEPS * 32 + lane;

    v8f acc = {};
    #pragma unroll 4
    for (int ks = 0; ks < KSTEPS; ++ks) {
        // A fragment (16-bit 16x32 layout): j<8 -> K=h*8+j ; j>=8 -> K=16+h*8+(j-8)
        const float* ap = arow + ks * 32;
        v16bf a;
        #pragma unroll
        for (int j = 0; j < 8; ++j) a[j] = (__bf16)ap[j];
        #pragma unroll
        for (int j = 0; j < 8; ++j) a[8 + j] = (__bf16)ap[16 + j];
        v16bf b = Bf[ks * 32];                    // 32B contiguous per lane
        acc = __builtin_amdgcn_wmma_f32_16x16x32_bf16(false, a, false, b,
                                                      (short)0, acc, false, false);
    }
    // C layout: VGPR r -> M = r + 8*h, N = m
    float* cp = C + (size_t)(rowBase + h * 8) * ldc + blockIdx.y * 16 + m;
    #pragma unroll
    for (int r = 0; r < 8; ++r) cp[(size_t)r * ldc] = acc[r];
}

// ---------------- batchnorm (training-mode batch stats, biased var) ----------------
__global__ void bn_stats_kernel(const float* __restrict__ y, const float* __restrict__ bias,
                                int Hc, float* __restrict__ sums, float* __restrict__ sumsq) {
    int ch = threadIdx.x;
    float b = bias[ch], s = 0.f, ss = 0.f;
    for (int r = blockIdx.x; r < N_; r += gridDim.x) {
        float t = y[(size_t)r * Hc + ch] + b;
        t = t > 0.f ? t : 0.f;
        s += t; ss += t * t;
    }
    atomicAdd(&sums[ch], s);
    atomicAdd(&sumsq[ch], ss);
}

__global__ void bn_finalize_kernel(const float* __restrict__ sums, const float* __restrict__ sumsq,
                                   const float* __restrict__ gamma, const float* __restrict__ beta,
                                   int Hc, float* __restrict__ coefA, float* __restrict__ coefC) {
    int ch = blockIdx.x * blockDim.x + threadIdx.x;
    if (ch >= Hc) return;
    float invN = 1.f / (float)N_;
    float mn = sums[ch] * invN;
    float vr = sumsq[ch] * invN - mn * mn;
    float a = gamma[ch] * rsqrtf(vr + EPS);
    coefA[ch] = a;
    coefC[ch] = beta[ch] - mn * a;
}

__global__ void bn_apply_kernel(const float* __restrict__ y, const float* __restrict__ bias,
                                const float* __restrict__ coefA, const float* __restrict__ coefC,
                                int Hc,
                                float* __restrict__ out1, int ld1, int off1,
                                float* __restrict__ out2, int ld2, int off2) {
    int idx = blockIdx.x * blockDim.x + threadIdx.x;
    if (idx >= N_ * Hc) return;
    int n = idx / Hc, ch = idx - n * Hc;
    float t = y[idx] + bias[ch];
    t = t > 0.f ? t : 0.f;
    float v = coefA[ch] * t + coefC[ch];
    out1[(size_t)n * ld1 + off1 + ch] = v;
    if (out2) out2[(size_t)n * ld2 + off2 + ch] = v;
}

// monotonic atomic float max (no NaNs expected)
__device__ __forceinline__ void atomicMaxF(float* addr, float v) {
    if (v >= 0.f) atomicMax((int*)addr, __float_as_int(v));
    else          atomicMin((unsigned int*)addr, __float_as_uint(v));
}

__global__ void pool_init_kernel(float* __restrict__ pooled) {
    int idx = blockIdx.x * blockDim.x + threadIdx.x;
    if (idx < G_ * 2 * H_) pooled[idx] = -3.402823466e+38f;
}

__global__ void bn_apply_pool_kernel(const float* __restrict__ y, const float* __restrict__ bias,
                                     const float* __restrict__ coefA, const float* __restrict__ coefC,
                                     const int* __restrict__ batch, float* __restrict__ pooled) {
    const int Hc = 2 * H_;
    int idx = blockIdx.x * blockDim.x + threadIdx.x;
    if (idx >= N_ * Hc) return;
    int n = idx / Hc, ch = idx - n * Hc;
    float t = y[idx] + bias[ch];
    t = t > 0.f ? t : 0.f;
    float v = coefA[ch] * t + coefC[ch];
    atomicMaxF(&pooled[(size_t)batch[n] * Hc + ch], v);
}

// ---------------- head: out = BN(relu(in @ W + b)) over G=64 rows ----------------
__global__ void head_linear_bn_kernel(const float* __restrict__ in, int Din,
                                      const float* __restrict__ W, const float* __restrict__ bias,
                                      const float* __restrict__ gamma, const float* __restrict__ beta,
                                      float* __restrict__ out, int Dout) {
    int c = blockIdx.x, g = threadIdx.x;
    float acc = bias[c];
    for (int i = 0; i < Din; ++i) acc += in[(size_t)g * Din + i] * W[(size_t)i * Dout + c];
    float t = acc > 0.f ? acc : 0.f;
    __shared__ float sh[64], sh2[64];
    sh[g] = t; sh2[g] = t * t;
    __syncthreads();
    for (int s = 32; s > 0; s >>= 1) {
        if (g < s) { sh[g] += sh[g + s]; sh2[g] += sh2[g + s]; }
        __syncthreads();
    }
    float mn = sh[0] * (1.f / 64.f);
    float vr = sh2[0] * (1.f / 64.f) - mn * mn;
    float a = gamma[c] * rsqrtf(vr + EPS);
    out[(size_t)g * Dout + c] = a * (t - mn) + beta[c];
}

// logits + log_softmax: grid = G blocks, block = 64 (first 40 lanes compute)
__global__ void logits_kernel(const float* __restrict__ f2, const float* __restrict__ W,
                              const float* __restrict__ b, float* __restrict__ out) {
    int g = blockIdx.x, c = threadIdx.x;
    float v = -3.402823466e+38f;
    if (c < C_) {
        float acc = b[c];
        for (int i = 0; i < H_; ++i) acc += f2[(size_t)g * H_ + i] * W[(size_t)i * C_ + c];
        v = acc;
    }
    __shared__ float red[64];
    red[c] = v; __syncthreads();
    for (int s = 32; s > 0; s >>= 1) { if (c < s) red[c] = fmaxf(red[c], red[c + s]); __syncthreads(); }
    float mx = red[0]; __syncthreads();
    float ex = (c < C_) ? expf(v - mx) : 0.f;
    red[c] = ex; __syncthreads();
    for (int s = 32; s > 0; s >>= 1) { if (c < s) red[c] += red[c + s]; __syncthreads(); }
    float lse = logf(red[0]) + mx;
    if (c < C_) out[(size_t)g * C_ + c] = v - lse;
}

// ---------------- launch ----------------
extern "C" void kernel_launch(void* const* d_in, const int* in_sizes, int n_in,
                              void* d_out, int out_size, void* d_ws, size_t ws_size,
                              hipStream_t stream) {
    (void)in_sizes; (void)n_in; (void)out_size; (void)ws_size;
    const float* x         = (const float*)d_in[0];
    const int*   eidx      = (const int*)d_in[1];
    const float* edge_attr = (const float*)d_in[2];
    const int*   batch     = (const int*)d_in[3];
    const float* W1 = (const float*)d_in[4],  *b1 = (const float*)d_in[5];
    const float* g1 = (const float*)d_in[6],  *be1 = (const float*)d_in[7];
    const float* W2 = (const float*)d_in[8],  *b2 = (const float*)d_in[9];
    const float* g2 = (const float*)d_in[10], *be2 = (const float*)d_in[11];
    const float* Wl = (const float*)d_in[12], *bl = (const float*)d_in[13];
    const float* gl = (const float*)d_in[14], *bel = (const float*)d_in[15];
    const float* Wf1 = (const float*)d_in[16], *bf1 = (const float*)d_in[17];
    const float* gf1 = (const float*)d_in[18], *bef1 = (const float*)d_in[19];
    const float* Wf2 = (const float*)d_in[20], *bf2 = (const float*)d_in[21];
    const float* gf2 = (const float*)d_in[22], *bef2 = (const float*)d_in[23];
    const float* Wf3 = (const float*)d_in[24], *bf3 = (const float*)d_in[25];

    const int* row = eidx;
    const int* col = eidx + E_;

    // workspace layout (floats; every region is a multiple of 8 floats -> 32B aligned)
    float* w = (float*)d_ws;
    float* norm   = w;                 w += E_;
    float* deg    = w;                 w += N_;
    float* xcat2  = w;                 w += (size_t)N_ * 512;   // conv2 hop concat; later reused as yl
    float* xcatL  = w;                 w += (size_t)N_ * 256;   // concat(x1, x2) for lin1
    float* ybuf   = w;                 w += (size_t)N_ * 128;   // y1 then y2
    float* xcat1  = w;                 w += (size_t)N_ * 64;    // conv1 hop concat
    float* sums   = w;                 w += 256;
    float* sumsq  = w;                 w += 256;
    float* coefA  = w;                 w += 256;
    float* coefC  = w;                 w += 256;
    float* pooled = w;                 w += G_ * 256;
    float* f1buf  = w;                 w += G_ * 256;
    float* f2buf  = w;                 w += G_ * 128;
    float* yl = xcat2;                 // alias: xcat2 dead before lin1 GEMM writes

    // bf16 pre-packed weight fragments (32B aligned: float region counts are all x8)
    __bf16* bw = (__bf16*)w;
    const int B1_TOT = (128 / 16) * (64 / 32)  * 512;   //  8192
    const int B2_TOT = (128 / 16) * (512 / 32) * 512;   // 65536
    const int BL_TOT = (256 / 16) * (256 / 32) * 512;   // 65536
    __bf16* W1f = bw;  bw += B1_TOT;
    __bf16* W2f = bw;  bw += B2_TOT;
    __bf16* Wlf = bw;  bw += BL_TOT;

    const int T = 256;
    auto cdiv = [](long long a, long long b) { return (int)((a + b - 1) / b); };

    // ---- graph norm + weight packing ----
    hipMemsetAsync(deg,   0, (size_t)N_ * 4,        stream);
    hipMemsetAsync(xcat1, 0, (size_t)N_ * 64 * 4,   stream);
    hipMemsetAsync(xcat2, 0, (size_t)N_ * 512 * 4,  stream);
    deg_kernel <<<cdiv(E_, T), T, 0, stream>>>(col, edge_attr, deg);
    norm_kernel<<<cdiv(E_, T), T, 0, stream>>>(row, col, edge_attr, deg, norm);
    pack_b_kernel<<<cdiv(B1_TOT, T), T, 0, stream>>>(W1, 128, 64 / 32,  W1f, B1_TOT);
    pack_b_kernel<<<cdiv(B2_TOT, T), T, 0, stream>>>(W2, 128, 512 / 32, W2f, B2_TOT);
    pack_b_kernel<<<cdiv(BL_TOT, T), T, 0, stream>>>(Wl, 256, 256 / 32, Wlf, BL_TOT);

    // ---- conv1: xcat1 = [x | A x | A^2 x | A^3 x]  (ld 64) ----
    copy_x_kernel<<<cdiv((long long)N_ * F_IN, T), T, 0, stream>>>(x, xcat1);
    for (int k = 1; k <= KHOP; ++k)
        propagate_kernel<<<cdiv((long long)E_ * 4, T), T, 0, stream>>>(
            xcat1, 64, (k - 1) * 16, xcat1, 64, k * 16, row, col, norm, 4);
    gemm_bf16_wmma<64><<<dim3(cdiv(N_ / 16, 8), 128 / 16), T, 0, stream>>>(
        xcat1, 64, W1f, ybuf, 128, N_);
    hipMemsetAsync(sums, 0, 512 * 4, stream);
    bn_stats_kernel   <<<512, 128, 0, stream>>>(ybuf, b1, 128, sums, sumsq);
    bn_finalize_kernel<<<1, 128, 0, stream>>>(sums, sumsq, g1, be1, 128, coefA, coefC);
    bn_apply_kernel<<<cdiv((long long)N_ * 128, T), T, 0, stream>>>(
        ybuf, b1, coefA, coefC, 128, xcat2, 512, 0, xcatL, 256, 0);

    // ---- conv2 hops inside xcat2 (ld 512) ----
    for (int k = 1; k <= KHOP; ++k)
        propagate_kernel<<<cdiv((long long)E_ * 32, T), T, 0, stream>>>(
            xcat2, 512, (k - 1) * 128, xcat2, 512, k * 128, row, col, norm, 32);
    gemm_bf16_wmma<512><<<dim3(cdiv(N_ / 16, 8), 128 / 16), T, 0, stream>>>(
        xcat2, 512, W2f, ybuf, 128, N_);
    hipMemsetAsync(sums, 0, 512 * 4, stream);
    bn_stats_kernel   <<<512, 128, 0, stream>>>(ybuf, b2, 128, sums, sumsq);
    bn_finalize_kernel<<<1, 128, 0, stream>>>(sums, sumsq, g2, be2, 128, coefA, coefC);
    bn_apply_kernel<<<cdiv((long long)N_ * 128, T), T, 0, stream>>>(
        ybuf, b2, coefA, coefC, 128, xcatL, 256, 128, nullptr, 0, 0);

    // ---- lin1: yl = xcatL @ Wl  [N,256] ----
    gemm_bf16_wmma<256><<<dim3(cdiv(N_ / 16, 8), 256 / 16), T, 0, stream>>>(
        xcatL, 256, Wlf, yl, 256, N_);
    hipMemsetAsync(sums, 0, 512 * 4, stream);
    bn_stats_kernel   <<<512, 256, 0, stream>>>(yl, bl, 256, sums, sumsq);
    bn_finalize_kernel<<<1, 256, 0, stream>>>(sums, sumsq, gl, bel, 256, coefA, coefC);
    pool_init_kernel<<<cdiv(G_ * 256, T), T, 0, stream>>>(pooled);
    bn_apply_pool_kernel<<<cdiv((long long)N_ * 256, T), T, 0, stream>>>(
        yl, bl, coefA, coefC, batch, pooled);

    // ---- head ----
    head_linear_bn_kernel<<<256, 64, 0, stream>>>(pooled, 256, Wf1, bf1, gf1, bef1, f1buf, 256);
    head_linear_bn_kernel<<<128, 64, 0, stream>>>(f1buf, 256, Wf2, bf2, gf2, bef2, f2buf, 128);
    logits_kernel<<<G_, 64, 0, stream>>>(f2buf, Wf3, bf3, (float*)d_out);
}
